// SimpleKeyboardGaAN_43061342109887
// MI455X (gfx1250) — matrix operand order repaired and checked by reading, not compiled, not run on hardware
//
#include <hip/hip_runtime.h>
#include <hip/hip_bf16.h>

typedef __bf16 bf16;
typedef __attribute__((ext_vector_type(16))) __bf16 v16bf;
typedef __attribute__((ext_vector_type(8)))  __bf16 v8bf;
typedef __attribute__((ext_vector_type(8)))  float  v8f;

#define N_NODES 50000
#define N_EDGES 800000
#define DIM_IN  768
#define HIDC    128
#define HEADS   4
#define NCLS    46

// ---------------------------------------------------------------------------
// elementwise helpers
// ---------------------------------------------------------------------------
__global__ void k_f2bf(const float* __restrict__ s, bf16* __restrict__ d, int n) {
  int i = blockIdx.x * blockDim.x + threadIdx.x;
  if (i < n) d[i] = (bf16)s[i];
}

__global__ void k_fill(float* __restrict__ p, float v, int n) {
  int i = blockIdx.x * blockDim.x + threadIdx.x;
  if (i < n) p[i] = v;
}

// Pack weight B[K,Ns] (f32, row-major) into WMMA B-fragment order (bf16),
// with the column dimension zero-padded to Np (multiple of 64):
//   Bp[(k>>5)*Np*32 + n*32 + (k&31)]  ;  n in [0,Np), zero for n >= Ns
// A lane then needs only two contiguous, always-valid 16-byte loads per frag.
__global__ void k_packB(const float* __restrict__ B, bf16* __restrict__ Bp,
                        int K, int Ns, int Np) {
  int i = blockIdx.x * blockDim.x + threadIdx.x;
  if (i >= K * Np) return;
  int k = i / Np, n = i - k * Np;
  bf16 v = (n < Ns) ? (bf16)B[(size_t)k * Ns + n] : (bf16)0.f;
  Bp[((size_t)(k >> 5) * Np + n) * 32 + (k & 31)] = v;
}

__device__ inline void atomicMaxF(float* a, float v) {
  if (v >= 0.f) atomicMax((int*)a, __float_as_int(v));
  else          atomicMin((unsigned int*)a, __float_as_uint(v));
}

// ---------------------------------------------------------------------------
// bf16 WMMA GEMM:  C[M,Ns] = opt_relu(A[M,K] @ B[K,Ns] + bias[Ns])
// A: row-major bf16. Bp: packed+padded fragment order (see k_packB).
// wave tile = 16 rows x 64 cols (1 A frag, 4 B frags, 4 wmma).
// block = 128 threads = 4 waves stacked in M -> 64x64 block tile.
// Requires M % 16 == 0 (true here: 50000 = 3125*16) -> wave-uniform M guard.
// No guards in the K loop: B is padded, so all loads are valid.
// ---------------------------------------------------------------------------
__global__ void k_wmma_gemm(const bf16* __restrict__ A, const bf16* __restrict__ Bp,
                            const float* __restrict__ bias, float* __restrict__ C,
                            bf16* __restrict__ Cbf, int M, int K, int Ns, int Np,
                            int relu)
{
  const int lane  = threadIdx.x & 31;
  const int wave  = threadIdx.x >> 5;
  const int mBase = blockIdx.x * 64 + wave * 16;
  if (mBase >= M) return;                    // wave-uniform: M % 16 == 0
  const int nBase = blockIdx.y * 64;
  const int row   = mBase + (lane & 15);
  const int klo   = (lane < 16) ? 0 : 8;     // A K-offset for this half-wave
  const int kbB   = (lane < 16) ? 0 : 8;     // B K-offset for this half-wave
  const int nlane = lane & 15;

  const bf16* ap  = A  + (size_t)row * K + klo;
  const bf16* bp0 = Bp + ((size_t)(nBase + nlane) * 32 + kbB);
  const size_t bstep = (size_t)Np * 32;      // packed elems per 32-K tile

  v8f acc[4];
  #pragma unroll
  for (int t = 0; t < 4; ++t)
    #pragma unroll
    for (int r = 0; r < 8; ++r) acc[t][r] = 0.f;

  for (int kb = 0; kb < K; kb += 32) {
    union { v16bf v; v8bf h[2]; } af, b0, b1, b2, b3;
    // issue all 10 b128 loads up front -> one clause, partial waits per wmma
    af.h[0] = *(const v8bf*)(ap);            // K = kb+klo+0..7
    af.h[1] = *(const v8bf*)(ap + 16);       // K = kb+klo+16..23
    b0.h[0] = *(const v8bf*)(bp0);
    b0.h[1] = *(const v8bf*)(bp0 + 16);
    b1.h[0] = *(const v8bf*)(bp0 + 16 * 32);
    b1.h[1] = *(const v8bf*)(bp0 + 16 * 32 + 16);
    b2.h[0] = *(const v8bf*)(bp0 + 32 * 32);
    b2.h[1] = *(const v8bf*)(bp0 + 32 * 32 + 16);
    b3.h[0] = *(const v8bf*)(bp0 + 48 * 32);
    b3.h[1] = *(const v8bf*)(bp0 + 48 * 32 + 16);
    __builtin_prefetch(ap + 64, 0, 1);       // global_prefetch_b8 down K stream

    acc[0] = __builtin_amdgcn_wmma_f32_16x16x32_bf16(
                 false, af.v, false, b0.v, (short)0, acc[0], false, false);
    acc[1] = __builtin_amdgcn_wmma_f32_16x16x32_bf16(
                 false, af.v, false, b1.v, (short)0, acc[1], false, false);
    acc[2] = __builtin_amdgcn_wmma_f32_16x16x32_bf16(
                 false, af.v, false, b2.v, (short)0, acc[2], false, false);
    acc[3] = __builtin_amdgcn_wmma_f32_16x16x32_bf16(
                 false, af.v, false, b3.v, (short)0, acc[3], false, false);

    ap  += 32;
    bp0 += bstep;
  }

  // epilogue: lane holds col nlane, rows r + (lane<16?0:8)
  const int rHi = (lane < 16) ? 0 : 8;
  #pragma unroll
  for (int t = 0; t < 4; ++t) {
    const int n = nBase + t * 16 + nlane;
    if (n >= Ns) continue;                   // only live for the N=46 GEMM
    const float bv = bias[n];
    #pragma unroll
    for (int r = 0; r < 8; ++r) {
      const int m = mBase + rHi + r;
      float v = acc[t][r] + bv;
      if (relu) v = fmaxf(v, 0.f);
      C[(size_t)m * Ns + n] = v;
      if (Cbf) Cbf[(size_t)m * Ns + n] = (bf16)v;
    }
  }
}

// ---------------------------------------------------------------------------
// conv1 edge kernels (4 heads x 128 ch). One wave32 per edge.
// ---------------------------------------------------------------------------
__global__ void k_logits1(const float* __restrict__ xl, const float* __restrict__ xr,
                          const int* __restrict__ ei, const float* __restrict__ eattr,
                          const float* __restrict__ W1e, const float* __restrict__ att1,
                          float* __restrict__ alog, float* __restrict__ amax)
{
  const int lane = threadIdx.x & 31;
  const int e = blockIdx.x * (blockDim.x >> 5) + (threadIdx.x >> 5);
  if (e >= N_EDGES) return;
  const int s = ei[e], d = ei[N_EDGES + e];
  const float ea = eattr[e];
  const float* pl = xl + (size_t)s * (HEADS * HIDC);
  const float* pr = xr + (size_t)d * (HEADS * HIDC);
  #pragma unroll
  for (int h = 0; h < HEADS; ++h) {
    float sum = 0.f;
    #pragma unroll
    for (int j = 0; j < 4; ++j) {
      const int c = h * HIDC + j * 32 + lane;
      float m = pl[c] + pr[c] + ea * W1e[c];
      m = (m > 0.f) ? m : 0.2f * m;                 // leaky_relu 0.2
      sum += m * att1[c];
    }
    #pragma unroll
    for (int off = 16; off; off >>= 1) sum += __shfl_xor(sum, off, 32);
    if (lane == 0) {
      alog[(size_t)e * HEADS + h] = sum;
      atomicMaxF(&amax[(size_t)d * HEADS + h], sum);
    }
  }
}

__global__ void k_exp1(const int* __restrict__ ei, float* __restrict__ alog,
                       const float* __restrict__ amax, float* __restrict__ denom)
{
  const int t = blockIdx.x * blockDim.x + threadIdx.x;
  if (t >= N_EDGES * HEADS) return;
  const int e = t >> 2, h = t & 3;
  const int d = ei[N_EDGES + e];
  const float a = __expf(alog[t] - amax[(size_t)d * HEADS + h]);
  alog[t] = a;
  atomicAdd(&denom[(size_t)d * HEADS + h], a);
}

__global__ void k_aggr1(const float* __restrict__ xl, const int* __restrict__ ei,
                        const float* __restrict__ alog, const float* __restrict__ denom,
                        float* __restrict__ out)
{
  const int lane = threadIdx.x & 31;
  const int e = blockIdx.x * (blockDim.x >> 5) + (threadIdx.x >> 5);
  if (e >= N_EDGES) return;
  const int s = ei[e], d = ei[N_EDGES + e];
  float alpha[HEADS];
  #pragma unroll
  for (int h = 0; h < HEADS; ++h)
    alpha[h] = alog[(size_t)e * HEADS + h] / (denom[(size_t)d * HEADS + h] + 1e-16f);
  const float* pl = xl + (size_t)s * (HEADS * HIDC);
  float* po = out + (size_t)d * (HEADS * HIDC);
  #pragma unroll
  for (int h = 0; h < HEADS; ++h)
    #pragma unroll
    for (int j = 0; j < 4; ++j) {
      const int c = h * HIDC + j * 32 + lane;
      atomicAdd(&po[c], alpha[h] * pl[c]);   // lands in 192MB L2 (buffer = 102MB)
    }
}

__global__ void k_finalize(float* __restrict__ h, const float* __restrict__ bias,
                           bf16* __restrict__ hbf, int n, int cols)
{
  const int i = blockIdx.x * blockDim.x + threadIdx.x;
  if (i >= n) return;
  float v = h[i] + bias[i % cols];
  v = fmaxf(v, 0.f);
  h[i] = v;
  hbf[i] = (bf16)v;
}

// ---------------------------------------------------------------------------
// conv2 edge kernels (1 head x 128 ch). One wave32 per edge.
// ---------------------------------------------------------------------------
__global__ void k_logits2(const float* __restrict__ hl, const float* __restrict__ hr,
                          const int* __restrict__ ei, const float* __restrict__ eattr,
                          const float* __restrict__ W2e, const float* __restrict__ att2,
                          float* __restrict__ alog, float* __restrict__ amax)
{
  const int lane = threadIdx.x & 31;
  const int e = blockIdx.x * (blockDim.x >> 5) + (threadIdx.x >> 5);
  if (e >= N_EDGES) return;
  const int s = ei[e], d = ei[N_EDGES + e];
  const float ea = eattr[e];
  const float* pl = hl + (size_t)s * HIDC;
  const float* pr = hr + (size_t)d * HIDC;
  float sum = 0.f;
  #pragma unroll
  for (int j = 0; j < 4; ++j) {
    const int c = j * 32 + lane;
    float m = pl[c] + pr[c] + ea * W2e[c];
    m = (m > 0.f) ? m : 0.2f * m;
    sum += m * att2[c];
  }
  #pragma unroll
  for (int off = 16; off; off >>= 1) sum += __shfl_xor(sum, off, 32);
  if (lane == 0) { alog[e] = sum; atomicMaxF(&amax[d], sum); }
}

__global__ void k_exp2(const int* __restrict__ ei, float* __restrict__ alog,
                       const float* __restrict__ amax, float* __restrict__ denom)
{
  const int e = blockIdx.x * blockDim.x + threadIdx.x;
  if (e >= N_EDGES) return;
  const int d = ei[N_EDGES + e];
  const float a = __expf(alog[e] - amax[d]);
  alog[e] = a;
  atomicAdd(&denom[d], a);
}

__global__ void k_aggr2(const float* __restrict__ hl, const int* __restrict__ ei,
                        const float* __restrict__ alog, const float* __restrict__ denom,
                        float* __restrict__ out)
{
  const int lane = threadIdx.x & 31;
  const int e = blockIdx.x * (blockDim.x >> 5) + (threadIdx.x >> 5);
  if (e >= N_EDGES) return;
  const int s = ei[e], d = ei[N_EDGES + e];
  const float alpha = alog[e] / (denom[d] + 1e-16f);
  const float* pl = hl + (size_t)s * HIDC;
  float* po = out + (size_t)d * HIDC;
  #pragma unroll
  for (int j = 0; j < 4; ++j) {
    const int c = j * 32 + lane;
    atomicAdd(&po[c], alpha * pl[c]);
  }
}

// ---------------------------------------------------------------------------
// host-side orchestration
// ---------------------------------------------------------------------------
extern "C" void kernel_launch(void* const* d_in, const int* in_sizes, int n_in,
                              void* d_out, int out_size, void* d_ws, size_t ws_size,
                              hipStream_t stream)
{
  const float* x     = (const float*)d_in[0];
  const int*   ei    = (const int*)  d_in[1];
  const float* eattr = (const float*)d_in[2];
  const float* W1l   = (const float*)d_in[3];
  const float* b1l   = (const float*)d_in[4];
  const float* W1r   = (const float*)d_in[5];
  const float* b1r   = (const float*)d_in[6];
  const float* W1e   = (const float*)d_in[7];
  const float* att1  = (const float*)d_in[8];
  const float* bias1 = (const float*)d_in[9];
  const float* W2l   = (const float*)d_in[10];
  const float* b2l   = (const float*)d_in[11];
  const float* W2r   = (const float*)d_in[12];
  const float* b2r   = (const float*)d_in[13];
  const float* W2e   = (const float*)d_in[14];
  const float* att2  = (const float*)d_in[15];
  const float* bias2 = (const float*)d_in[16];
  const float* fcW1  = (const float*)d_in[17];
  const float* fcb1  = (const float*)d_in[18];
  const float* fcW2  = (const float*)d_in[19];
  const float* fcb2  = (const float*)d_in[20];
  float* out = (float*)d_out;

  const int Nn = N_NODES, Ee = N_EDGES;
  const int C1 = HEADS * HIDC;          // 512
  const int NCLSP = 64;                 // NCLS=46 padded to 64

  size_t off = 0;
  auto alloc = [&](size_t bytes) -> char* {
    char* p = (char*)d_ws + off;
    off += (bytes + 255) & ~(size_t)255;
    return p;
  };
  bf16*  xbf   = (bf16*) alloc((size_t)Nn * DIM_IN * 2);
  bf16*  w1lb  = (bf16*) alloc((size_t)DIM_IN * C1 * 2);
  bf16*  w1rb  = (bf16*) alloc((size_t)DIM_IN * C1 * 2);
  float* xl    = (float*)alloc((size_t)Nn * C1 * 4);
  float* xr    = (float*)alloc((size_t)Nn * C1 * 4);
  float* alog1 = (float*)alloc((size_t)Ee * HEADS * 4);
  float* amax1 = (float*)alloc((size_t)Nn * HEADS * 4);
  float* den1  = (float*)alloc((size_t)Nn * HEADS * 4);
  float* h1    = (float*)alloc((size_t)Nn * C1 * 4);
  bf16*  h1bf  = (bf16*) alloc((size_t)Nn * C1 * 2);
  bf16*  w2lb  = (bf16*) alloc((size_t)C1 * HIDC * 2);
  bf16*  w2rb  = (bf16*) alloc((size_t)C1 * HIDC * 2);
  float* hl    = (float*)alloc((size_t)Nn * HIDC * 4);
  float* hr    = (float*)alloc((size_t)Nn * HIDC * 4);
  float* alog2 = (float*)alloc((size_t)Ee * 4);
  float* amax2 = (float*)alloc((size_t)Nn * 4);
  float* den2  = (float*)alloc((size_t)Nn * 4);
  float* h2    = (float*)alloc((size_t)Nn * HIDC * 4);
  bf16*  h2bf  = (bf16*) alloc((size_t)Nn * HIDC * 2);
  bf16*  fw1b  = (bf16*) alloc((size_t)HIDC * 64 * 2);
  float* fc1   = (float*)alloc((size_t)Nn * 64 * 4);
  bf16*  fc1bf = (bf16*) alloc((size_t)Nn * 64 * 2);
  bf16*  fw2b  = (bf16*) alloc((size_t)64 * NCLSP * 2);

  auto cvt = [&](const float* s, bf16* d, int n) {
    k_f2bf<<<(n + 255) / 256, 256, 0, stream>>>(s, d, n);
  };
  auto pack = [&](const float* s, bf16* d, int K, int Ns, int Np) {
    k_packB<<<(K * Np + 255) / 256, 256, 0, stream>>>(s, d, K, Ns, Np);
  };
  auto fill = [&](float* p, float v, int n) {
    k_fill<<<(n + 255) / 256, 256, 0, stream>>>(p, v, n);
  };

  // ---- conv1 dense transforms (bf16 WMMA) ----
  cvt(x, xbf, Nn * DIM_IN);
  pack(W1l, w1lb, DIM_IN, C1, C1);
  pack(W1r, w1rb, DIM_IN, C1, C1);
  {
    dim3 g((Nn + 63) / 64, C1 / 64), b(128);
    k_wmma_gemm<<<g, b, 0, stream>>>(xbf, w1lb, b1l, xl, nullptr, Nn, DIM_IN, C1, C1, 0);
    k_wmma_gemm<<<g, b, 0, stream>>>(xbf, w1rb, b1r, xr, nullptr, Nn, DIM_IN, C1, C1, 0);
  }

  // ---- conv1 edge softmax + aggregation ----
  fill(amax1, -3.0e38f, Nn * HEADS);
  fill(den1, 0.f, Nn * HEADS);
  fill(h1, 0.f, Nn * C1);
  const int EB = (Ee + 7) / 8;  // 8 waves (256 threads) per block, wave per edge
  k_logits1<<<EB, 256, 0, stream>>>(xl, xr, ei, eattr, W1e, att1, alog1, amax1);
  k_exp1<<<(Ee * HEADS + 255) / 256, 256, 0, stream>>>(ei, alog1, amax1, den1);
  k_aggr1<<<EB, 256, 0, stream>>>(xl, ei, alog1, den1, h1);
  k_finalize<<<(Nn * C1 + 255) / 256, 256, 0, stream>>>(h1, bias1, h1bf, Nn * C1, C1);

  // ---- conv2 dense transforms ----
  pack(W2l, w2lb, C1, HIDC, HIDC);
  pack(W2r, w2rb, C1, HIDC, HIDC);
  {
    dim3 g((Nn + 63) / 64, HIDC / 64), b(128);
    k_wmma_gemm<<<g, b, 0, stream>>>(h1bf, w2lb, b2l, hl, nullptr, Nn, C1, HIDC, HIDC, 0);
    k_wmma_gemm<<<g, b, 0, stream>>>(h1bf, w2rb, b2r, hr, nullptr, Nn, C1, HIDC, HIDC, 0);
  }

  // ---- conv2 edge softmax + aggregation (1 head) ----
  fill(amax2, -3.0e38f, Nn);
  fill(den2, 0.f, Nn);
  fill(h2, 0.f, Nn * HIDC);
  k_logits2<<<EB, 256, 0, stream>>>(hl, hr, ei, eattr, W2e, att2, alog2, amax2);
  k_exp2<<<(Ee + 255) / 256, 256, 0, stream>>>(ei, alog2, amax2, den2);
  k_aggr2<<<EB, 256, 0, stream>>>(hl, ei, alog2, den2, h2);
  // heads==1 => mean == identity; bias + relu + bf16
  k_finalize<<<(Nn * HIDC + 255) / 256, 256, 0, stream>>>(h2, bias2, h2bf, Nn * HIDC, HIDC);

  // ---- FC head (WMMA) ----
  pack(fcW1, fw1b, HIDC, 64, 64);
  pack(fcW2, fw2b, 64, NCLS, NCLSP);
  {
    dim3 g1((Nn + 63) / 64, 1), b(128);
    k_wmma_gemm<<<g1, b, 0, stream>>>(h2bf, fw1b, fcb1, fc1, fc1bf, Nn, HIDC, 64, 64, 1);
    k_wmma_gemm<<<g1, b, 0, stream>>>(fc1bf, fw2b, fcb2, out, nullptr, Nn, 64, NCLS, NCLSP, 0);
  }
}
